// Instant_forecasting_64072322121857
// MI455X (gfx1250) — matrix operand order repaired and checked by reading, not compiled, not run on hardware
//
#include <hip/hip_runtime.h>
#include <stdint.h>
#include <stddef.h>

#define B_      32
#define V_      512
#define T_      12
#define FIN_    2
#define DG_     20
#define CH_     16
#define HOR_    12
#define KS_     2
#define RF_     13
#define NL_     8
#define EPS_    0.3f
#define BN_EPS_ 1e-5f

typedef float    v8f   __attribute__((ext_vector_type(8)));
typedef __bf16   v16bf __attribute__((ext_vector_type(16)));
typedef uint32_t u32x4 __attribute__((ext_vector_type(4)));

union FragU { v16bf v; u32x4 q[2]; };

__device__ __forceinline__ float sigm(float x) { return 1.0f / (1.0f + expf(-x)); }

__device__ __forceinline__ uint16_t f32_to_bf16(float f) {
  union { float f; uint32_t u; } a; a.f = f;
  uint32_t r = (a.u + 0x7FFFu + ((a.u >> 16) & 1u)) >> 16;
  return (uint16_t)r;
}
__device__ __forceinline__ float bf16_to_f32(uint16_t h) {
  union { uint32_t u; float f; } a; a.u = ((uint32_t)h) << 16;
  return a.f;
}

// ---------------- graph learner ----------------

// CR[d][n] = sum_j linW[d][j] cos(2pi j n / 24); CI = -sum sin. costab[q]=cos(2pi q/512)
__global__ void k_tables(const float* __restrict__ linW, float* __restrict__ CR,
                         float* __restrict__ CI, float* __restrict__ ct) {
  int id = threadIdx.x;                 // 1 block x 512
  ct[id] = cosf(6.283185307179586f * (float)id / 512.0f);
  if (id < DG_ * 24) {
    int d = id / 24, n = id % 24;
    float sr = 0.f, si = 0.f;
    for (int j = 0; j < 24; ++j) {
      float th = 6.283185307179586f * (float)(j * n) / 24.0f;
      float w = linW[d * 24 + j];
      sr += w * cosf(th);
      si -= w * sinf(th);
    }
    CR[id] = sr; CI[id] = si;
  }
}

// per (b,v): DFT(24) folded with lin projection -> re/im/Am/S [DG]
__global__ void k_dft(const float* __restrict__ input, const float* __restrict__ CR,
                      const float* __restrict__ CI, const float* __restrict__ linb,
                      float* __restrict__ re, float* __restrict__ im,
                      float* __restrict__ Am, float* __restrict__ Sm) {
  int id = blockIdx.x * blockDim.x + threadIdx.x;
  if (id >= B_ * V_) return;
  int b = id / V_, v = id % V_;
  float xr[24];
  for (int j = 0; j < 24; ++j) {
    int t = j >> 1, f = j & 1;
    xr[j] = input[(((size_t)t * B_ + b) * V_ + v) * FIN_ + f];
  }
  for (int d = 0; d < DG_; ++d) {
    float sr = linb[d], si = linb[d];
    for (int n = 0; n < 24; ++n) { sr += xr[n] * CR[d * 24 + n]; si += xr[n] * CI[d * 24 + n]; }
    size_t o = (size_t)id * DG_ + d;
    re[o] = sr; im[o] = si;
    Am[o] = sqrtf(sr * sr + si * si);
    Sm[o] = atanf(sr / (si + 1e-4f));
  }
}

// irfft(n=512) along node axis per (b,d); spectrum cached in LDS
__global__ void k_irfft(const float* __restrict__ re, const float* __restrict__ im,
                        const float* __restrict__ ct, float* __restrict__ iff) {
  __shared__ float rs[257], is[257], cs[512];
  int bd = blockIdx.x;
  int b = bd / DG_, d = bd % DG_;
  int v = threadIdx.x;                  // 512 threads
  for (int k = v; k < 257; k += 512) {
    rs[k] = re[((size_t)b * V_ + k) * DG_ + d];
    is[k] = im[((size_t)b * V_ + k) * DG_ + d];
  }
  cs[v] = ct[v];
  __syncthreads();
  float acc = rs[0] + ((v & 1) ? -rs[256] : rs[256]);
  for (int k = 1; k < 256; ++k) {
    int q = (k * v) & 511;
    acc += 2.0f * (rs[k] * cs[q] - is[k] * cs[(q + 384) & 511]);   // sin = cos(th - pi/2)
  }
  iff[((size_t)b * V_ + v) * DG_ + d] = acc * (1.0f / 512.0f);
}

// adj[v][w] = sigmoid(mean_b sigmoid((iff.fc1 + Am.fc2 + S.fc3 + biases)/3))
__global__ void k_adj(const float* __restrict__ iff, const float* __restrict__ Am,
                      const float* __restrict__ Sm,
                      const float* __restrict__ f1W, const float* __restrict__ f1b,
                      const float* __restrict__ f2W, const float* __restrict__ f2b,
                      const float* __restrict__ f3W, const float* __restrict__ f3b,
                      float* __restrict__ adj) {
  int v = blockIdx.x >> 1;
  int w = ((blockIdx.x & 1) << 8) + threadIdx.x;
  float w1[DG_], w2[DG_], w3[DG_];
  for (int d = 0; d < DG_; ++d) {
    w1[d] = f1W[w * DG_ + d]; w2[d] = f2W[w * DG_ + d]; w3[d] = f3W[w * DG_ + d];
  }
  float bias = f1b[w] + f2b[w] + f3b[w];
  float acc = 0.f;
  for (int b = 0; b < B_; ++b) {
    const float* pi = iff + ((size_t)b * V_ + v) * DG_;
    const float* pa = Am  + ((size_t)b * V_ + v) * DG_;
    const float* ps = Sm  + ((size_t)b * V_ + v) * DG_;
    float s = bias;
    for (int d = 0; d < DG_; ++d) s += pi[d] * w1[d] + pa[d] * w2[d] + ps[d] * w3[d];
    acc += sigm(s * (1.0f / 3.0f));
  }
  adj[(size_t)v * V_ + w] = sigm(acc * (1.0f / (float)B_));
}

// pack Lm/Hm (bf16) into this code's WMMA B-operand order (contiguous k, same as the A loads):
// tile(kt,nt): lane<16 -> col nt*16+lane, K = kt*32 + 0..15 ; lane>=16 -> K = kt*32 + 16..31
__global__ void k_pack(const float* __restrict__ adj, uint16_t* __restrict__ PL,
                       uint16_t* __restrict__ PH) {
  int id = blockIdx.x * blockDim.x + threadIdx.x;   // [2][16][32][32]
  if (id >= 2 * 16 * 32 * 32) return;
  int lane = id & 31; int t = id >> 5;
  int nt = t & 31; t >>= 5;
  int kt = t & 15; t >>= 4;
  int mat = t;
  int w = nt * 16 + (lane & 15);
  int vbase = kt * 32 + ((lane >> 4) * 16);
  uint16_t* out = (mat ? PH : PL) + (((size_t)(kt * 32 + nt)) * 32 + lane) * 16;
  for (int j = 0; j < 16; ++j) {
    int v = vbase + j;
    float a = adj[(size_t)v * V_ + w];
    float val = mat ? -a : a;
    if (v == w) val += EPS_;
    out[j] = f32_to_bf16(val);
  }
}

// ---------------- trunk ----------------

// pad(RF-T) + 1x1 start conv -> x [B,CH,V,13]
__global__ void k_xstart(const float* __restrict__ input, const float* __restrict__ sW,
                         const float* __restrict__ sb, float* __restrict__ x) {
  int id = blockIdx.x * blockDim.x + threadIdx.x;
  if (id >= B_ * CH_ * V_ * RF_) return;
  int t = id % RF_; int r = id / RF_;
  int v = r % V_; r /= V_;
  int c = r % CH_; int b = r / CH_;
  float acc = sb[c];
  if (t > 0) {
    int tt = t - 1;                      // 1 zero pad on the left
    for (int f = 0; f < FIN_; ++f)
      acc += sW[c * FIN_ + f] * input[(((size_t)tt * B_ + b) * V_ + v) * FIN_ + f];
  }
  x[id] = acc;
}

// dilated K=2 gated conv; writes bf16 activations row-major [m=(b,c,t)][v] for WMMA A-operand
__global__ void k_tconv(const float* __restrict__ xin, const float* __restrict__ fW,
                        const float* __restrict__ fb, const float* __restrict__ gW,
                        const float* __restrict__ gb, uint16_t* __restrict__ Xt,
                        int Lc, int dil, int layer) {
  int id = blockIdx.x * blockDim.x + threadIdx.x;
  if (id >= B_ * CH_ * V_ * Lc) return;
  int t = id % Lc; int r = id / Lc;
  int v = r % V_; r /= V_;
  int co = r % CH_; int b = r / CH_;
  const float* Wf = fW + ((size_t)layer * CH_ + co) * CH_ * KS_;
  const float* Wg = gW + ((size_t)layer * CH_ + co) * CH_ * KS_;
  float f = fb[layer * CH_ + co], g = gb[layer * CH_ + co];
  for (int ci = 0; ci < CH_; ++ci) {
    const float* xp = xin + (((size_t)b * CH_ + ci) * V_ + v) * RF_ + t;
    float x0 = xp[0], x1 = xp[dil];
    f += Wf[ci * KS_ + 0] * x0 + Wf[ci * KS_ + 1] * x1;
    g += Wg[ci * KS_ + 0] * x0 + Wg[ci * KS_ + 1] * x1;
  }
  float gx = tanhf(f) * sigm(g);
  Xt[(((size_t)b * CH_ + co) * Lc + t) * V_ + v] = f32_to_bf16(gx);
}

// skip 1x1 conv, accumulated right-aligned into skip[B,CH,V,12]
__global__ void k_skip(const uint16_t* __restrict__ Xt, const float* __restrict__ sW,
                       const float* __restrict__ sb, float* __restrict__ skip,
                       int Lc, int layer) {
  int id = blockIdx.x * blockDim.x + threadIdx.x;
  if (id >= B_ * CH_ * V_ * Lc) return;
  int t = id % Lc; int r = id / Lc;
  int v = r % V_; r /= V_;
  int co = r % CH_; int b = r / CH_;
  float s = sb[layer * CH_ + co];
  for (int ci = 0; ci < CH_; ++ci)
    s += sW[((size_t)layer * CH_ + co) * CH_ + ci] *
         bf16_to_f32(Xt[(((size_t)b * CH_ + ci) * Lc + t) * V_ + v]);
  skip[(((size_t)b * CH_ + co) * V_ + v) * T_ + (T_ - Lc + t)] += s;
}

// Y[m,w] = sum_v X[m,v] * A[v,w], bf16 in / f32 acc / bf16 out.
// gridDim.y: 0 -> X0*PL->Y0, 1 -> X1*PH->Y1.
// One wave per 16x64 output strip: 4 independent accumulators, A-fragment reused 4x,
// 64 WMMAs per wave over K=512.
__global__ void k_gemm(const uint16_t* __restrict__ X0, const uint16_t* __restrict__ X1,
                       const uint16_t* __restrict__ PL, const uint16_t* __restrict__ PH,
                       uint16_t* __restrict__ Y0, uint16_t* __restrict__ Y1, int M) {
  const uint16_t* X = blockIdx.y ? X1 : X0;
  const uint16_t* P = blockIdx.y ? PH : PL;
  uint16_t*       Y = blockIdx.y ? Y1 : Y0;
  int wave = (int)((blockIdx.x * blockDim.x + threadIdx.x) >> 5);
  int lane = threadIdx.x & 31;
  const int tilesN = V_ / 16;                       // 32 n-tiles
  const int groups = tilesN / 4;                    // 8 strips of 4 n-tiles
  int strips = (M / 16) * groups;
  if (wave >= strips) return;                       // wave-uniform: EXEC stays all-ones
  int m0 = (wave / groups) * 16;
  int ng = wave % groups;                           // which 4-tile strip
  int half = lane >> 4;
  const uint16_t* xrow = X + (size_t)(m0 + (lane & 15)) * V_;
  v8f acc0 = {0.f, 0.f, 0.f, 0.f, 0.f, 0.f, 0.f, 0.f};
  v8f acc1 = acc0, acc2 = acc0, acc3 = acc0;
#pragma unroll
  for (int kt = 0; kt < V_ / 32; ++kt) {
    int k0 = kt * 32;
    FragU a;
    // A-operand (16x32 bf16): lo lanes K 0-15, hi lanes K 16-31
    a.q[0] = *(const u32x4*)(xrow + k0 + half * 16);
    a.q[1] = *(const u32x4*)(xrow + k0 + half * 16 + 8);
    // B-operands: pre-packed, 32 contiguous bytes/lane; consecutive n-tiles 512 elems apart
    const uint16_t* bp = P + (((size_t)(kt * tilesN + ng * 4)) * 32 + lane) * 16;
    FragU b0, b1, b2, b3;
    b0.q[0] = *(const u32x4*)(bp +    0); b0.q[1] = *(const u32x4*)(bp +    8);
    b1.q[0] = *(const u32x4*)(bp +  512); b1.q[1] = *(const u32x4*)(bp +  520);
    b2.q[0] = *(const u32x4*)(bp + 1024); b2.q[1] = *(const u32x4*)(bp + 1032);
    b3.q[0] = *(const u32x4*)(bp + 1536); b3.q[1] = *(const u32x4*)(bp + 1544);
    acc0 = __builtin_amdgcn_wmma_f32_16x16x32_bf16(false, a.v, false, b0.v,
                                                   (short)0, acc0, false, false);
    acc1 = __builtin_amdgcn_wmma_f32_16x16x32_bf16(false, a.v, false, b1.v,
                                                   (short)0, acc1, false, false);
    acc2 = __builtin_amdgcn_wmma_f32_16x16x32_bf16(false, a.v, false, b2.v,
                                                   (short)0, acc2, false, false);
    acc3 = __builtin_amdgcn_wmma_f32_16x16x32_bf16(false, a.v, false, b3.v,
                                                   (short)0, acc3, false, false);
  }
  int rbase = m0 + half * 8;                        // D: vgpr r -> row r (lo) / r+8 (hi)
  int c0 = (ng * 4) * 16 + (lane & 15);
#pragma unroll
  for (int r = 0; r < 8; ++r) {
    size_t row = (size_t)(rbase + r) * V_;
    Y[row + c0 +  0] = f32_to_bf16(acc0[r]);
    Y[row + c0 + 16] = f32_to_bf16(acc1[r]);
    Y[row + c0 + 32] = f32_to_bf16(acc2[r]);
    Y[row + c0 + 48] = f32_to_bf16(acc3[r]);
  }
}

// gconv(64->16) over [Y1(L),Y2(LL),Y3(H),Y4(HH)] + residual + BN -> new x
__global__ void k_epi(const uint16_t* __restrict__ Y1, const uint16_t* __restrict__ Y2,
                      const uint16_t* __restrict__ Y3, const uint16_t* __restrict__ Y4,
                      const float* __restrict__ gW, const float* __restrict__ gb,
                      const float* __restrict__ xin, float* __restrict__ xout,
                      const float* __restrict__ bnG, const float* __restrict__ bnB,
                      const float* __restrict__ bnM, const float* __restrict__ bnV,
                      int Lc, int dil, int layer) {
  int id = blockIdx.x * blockDim.x + threadIdx.x;
  if (id >= B_ * CH_ * V_ * Lc) return;
  int t = id % Lc; int r = id / Lc;
  int w = r % V_; r /= V_;
  int co = r % CH_; int b = r / CH_;
  const float* W = gW + ((size_t)layer * CH_ + co) * (4 * CH_);
  float h = gb[layer * CH_ + co];
  const uint16_t* Ys[4] = {Y1, Y2, Y3, Y4};
#pragma unroll
  for (int g = 0; g < 4; ++g) {
    const uint16_t* Y = Ys[g];
    for (int cc = 0; cc < CH_; ++cc)
      h += W[g * CH_ + cc] * bf16_to_f32(Y[(((size_t)b * CH_ + cc) * Lc + t) * V_ + w]);
  }
  h += xin[(((size_t)b * CH_ + co) * V_ + w) * RF_ + (t + dil)];   // residual[..., -Lc:]
  int p = layer * CH_ + co;
  h = (h - bnM[p]) * rsqrtf(bnV[p] + BN_EPS_) * bnG[p] + bnB[p];
  xout[(((size_t)b * CH_ + co) * V_ + w) * RF_ + t] = h;
}

__global__ void k_zero(float* __restrict__ p, int n) {
  int id = blockIdx.x * blockDim.x + threadIdx.x;
  if (id < n) p[id] = 0.f;
}

// relu(skip) -> relu(end1) -> end2 ; out [B,HOR,V,1]
__global__ void k_head(const float* __restrict__ skip, const float* __restrict__ e1W,
                       const float* __restrict__ e1b, const float* __restrict__ e2W,
                       const float* __restrict__ e2b, float* __restrict__ out) {
  int id = blockIdx.x * blockDim.x + threadIdx.x;
  if (id >= B_ * V_) return;
  int b = id / V_, v = id % V_;
  float s[CH_], t1[CH_];
  for (int ci = 0; ci < CH_; ++ci) {
    float x = skip[(((size_t)b * CH_ + ci) * V_ + v) * T_ + (T_ - 1)];
    s[ci] = x > 0.f ? x : 0.f;
  }
  for (int co = 0; co < CH_; ++co) {
    float a = e1b[co];
    for (int ci = 0; ci < CH_; ++ci) a += e1W[co * CH_ + ci] * s[ci];
    t1[co] = a > 0.f ? a : 0.f;
  }
  for (int h = 0; h < HOR_; ++h) {
    float a = e2b[h];
    for (int co = 0; co < CH_; ++co) a += e2W[h * CH_ + co] * t1[co];
    out[((size_t)b * HOR_ + h) * V_ + v] = a;
  }
}

// ---------------- launcher ----------------

extern "C" void kernel_launch(void* const* d_in, const int* in_sizes, int n_in,
                              void* d_out, int out_size, void* d_ws, size_t ws_size,
                              hipStream_t stream) {
  (void)in_sizes; (void)n_in; (void)out_size; (void)ws_size;
  const float* input  = (const float*)d_in[0];
  const float* startW = (const float*)d_in[1];
  const float* startb = (const float*)d_in[2];
  const float* filtW  = (const float*)d_in[3];
  const float* filtb  = (const float*)d_in[4];
  const float* gateW  = (const float*)d_in[5];
  const float* gateb  = (const float*)d_in[6];
  const float* skipW  = (const float*)d_in[7];
  const float* skipb  = (const float*)d_in[8];
  const float* gconvW = (const float*)d_in[9];
  const float* gconvb = (const float*)d_in[10];
  const float* bnG    = (const float*)d_in[11];
  const float* bnB    = (const float*)d_in[12];
  const float* bnM    = (const float*)d_in[13];
  const float* bnV    = (const float*)d_in[14];
  const float* e1W    = (const float*)d_in[15];
  const float* e1b    = (const float*)d_in[16];
  const float* e2W    = (const float*)d_in[17];
  const float* e2b    = (const float*)d_in[18];
  const float* linW   = (const float*)d_in[19];
  const float* linb   = (const float*)d_in[20];
  const float* fc1W   = (const float*)d_in[21];
  const float* fc1b   = (const float*)d_in[22];
  const float* fc2W   = (const float*)d_in[23];
  const float* fc2b   = (const float*)d_in[24];
  const float* fc3W   = (const float*)d_in[25];
  const float* fc3b   = (const float*)d_in[26];

  float* out = (float*)d_out;
  float* adj = out + (size_t)B_ * HOR_ * V_;        // tuple tail: adj [V,V]

  char* wp = (char*)d_ws;
  auto alloc = [&](size_t bytes) -> void* {
    void* p = (void*)wp;
    wp += (bytes + 255) & ~(size_t)255;
    return p;
  };
  float* CR  = (float*)alloc(DG_ * 24 * sizeof(float));
  float* CI  = (float*)alloc(DG_ * 24 * sizeof(float));
  float* ct  = (float*)alloc(512 * sizeof(float));
  float* re  = (float*)alloc((size_t)B_ * V_ * DG_ * sizeof(float));
  float* im  = (float*)alloc((size_t)B_ * V_ * DG_ * sizeof(float));
  float* Am  = (float*)alloc((size_t)B_ * V_ * DG_ * sizeof(float));
  float* Sm  = (float*)alloc((size_t)B_ * V_ * DG_ * sizeof(float));
  float* iff = (float*)alloc((size_t)B_ * V_ * DG_ * sizeof(float));
  uint16_t* PL = (uint16_t*)alloc((size_t)V_ * V_ * sizeof(uint16_t));
  uint16_t* PH = (uint16_t*)alloc((size_t)V_ * V_ * sizeof(uint16_t));
  float* xA  = (float*)alloc((size_t)B_ * CH_ * V_ * RF_ * sizeof(float));
  float* xB  = (float*)alloc((size_t)B_ * CH_ * V_ * RF_ * sizeof(float));
  size_t ybytes = (size_t)B_ * CH_ * T_ * V_ * sizeof(uint16_t);   // M_max = 6144 rows
  uint16_t* Xt = (uint16_t*)alloc(ybytes);
  uint16_t* Y1 = (uint16_t*)alloc(ybytes);
  uint16_t* Y2 = (uint16_t*)alloc(ybytes);
  uint16_t* Y3 = (uint16_t*)alloc(ybytes);
  uint16_t* Y4 = (uint16_t*)alloc(ybytes);
  float* skip = (float*)alloc((size_t)B_ * CH_ * V_ * T_ * sizeof(float));

  // graph learner
  k_tables<<<1, 512, 0, stream>>>(linW, CR, CI, ct);
  k_dft<<<(B_ * V_ + 255) / 256, 256, 0, stream>>>(input, CR, CI, linb, re, im, Am, Sm);
  k_irfft<<<B_ * DG_, 512, 0, stream>>>(re, im, ct, iff);
  k_adj<<<V_ * 2, 256, 0, stream>>>(iff, Am, Sm, fc1W, fc1b, fc2W, fc2b, fc3W, fc3b, adj);
  k_pack<<<(2 * 16 * 32 * 32 + 255) / 256, 256, 0, stream>>>(adj, PL, PH);

  // trunk
  k_xstart<<<(B_ * CH_ * V_ * RF_ + 255) / 256, 256, 0, stream>>>(input, startW, startb, xA);
  k_zero<<<(B_ * CH_ * V_ * T_ + 255) / 256, 256, 0, stream>>>(skip, B_ * CH_ * V_ * T_);

  const int dils[NL_] = {1, 2, 1, 2, 1, 2, 1, 2};
  int L = RF_;
  float* xin = xA;
  float* xout = xB;
  for (int i = 0; i < NL_; ++i) {
    int dil = dils[i];
    int Lc = L - dil;
    int M = B_ * CH_ * Lc;
    int n1 = B_ * CH_ * V_ * Lc;
    k_tconv<<<(n1 + 255) / 256, 256, 0, stream>>>(xin, filtW, filtb, gateW, gateb, Xt, Lc, dil, i);
    k_skip<<<(n1 + 255) / 256, 256, 0, stream>>>(Xt, skipW, skipb, skip, Lc, i);
    int strips = (M / 16) * ((V_ / 16) / 4);        // one wave per 16x64 strip
    dim3 g((unsigned)((strips * 32 + 255) / 256), 2);
    k_gemm<<<g, 256, 0, stream>>>(Xt, Xt, PL, PH, Y1, Y3, M);   // hop 1: X*Lm, X*Hm
    k_gemm<<<g, 256, 0, stream>>>(Y1, Y3, PL, PH, Y2, Y4, M);   // hop 2: (X*Lm)*Lm, (X*Hm)*Hm
    k_epi<<<(n1 + 255) / 256, 256, 0, stream>>>(Y1, Y2, Y3, Y4, gconvW, gconvb, xin, xout,
                                                bnG, bnB, bnM, bnV, Lc, dil, i);
    float* tmp = xin; xin = xout; xout = tmp;
    L = Lc;
  }

  k_head<<<(B_ * V_ + 255) / 256, 256, 0, stream>>>(skip, e1W, e1b, e2W, e2b, out);
}
